// GuoUnpool_44160853737495
// MI455X (gfx1250) — compile-verified, matrix operand order checked by reading, not run on hardware
//
#include <hip/hip_runtime.h>

#define NN     20000
#define DX_    512
#define DW_    64
#define DY_    256
#define M_     320000
#define E2_    (2*M_)
#define DP_    384
#define DF_    832
#define HR_    256
#define HY_    128
#define EPS_   1e-9f
#define LNEPS_ 1e-5f
#define SLOPE_ 0.05f
#define NB256  ((NN + 255) / 256)

typedef __attribute__((ext_vector_type(16))) __bf16 v16bf;
typedef __attribute__((ext_vector_type(8)))  float  v8f;

__device__ __forceinline__ v8f wmma_bf16(v16bf a, v16bf b, v8f c) {
  return __builtin_amdgcn_wmma_f32_16x16x32_bf16(false, a, false, b, (short)0, c,
                                                 false, false);
}

// ---- CDNA5 async global->LDS copy (ASYNCcnt) --------------------------------
__device__ __forceinline__ void async_b128(unsigned lds_off, const void* g) {
  asm volatile("global_load_async_to_lds_b128 %0, %1, off"
               :: "v"(lds_off), "v"(g) : "memory");
}
__device__ __forceinline__ void wait_async_0() { asm volatile("s_wait_asynccnt 0x0" ::: "memory"); }

// generic shared pointer -> wave-relative LDS byte address (addr[31:0])
__device__ __forceinline__ unsigned ldsoff(const void* p) {
  return (unsigned)(unsigned long long)p;
}

// copy one contiguous swizzled weight panel into LDS (BYTES % 4096 == 0)
template<int BYTES>
__device__ __forceinline__ void issue_panel(unsigned lds_off, const char* g, int tid) {
  const int base = tid * 16;
#pragma unroll
  for (int o = 0; o < BYTES; o += 4096) async_b128(lds_off + base + o, g + base + o);
}

__device__ __forceinline__ unsigned pk_bf16(float a, float b) {
  union { __bf16 h[2]; unsigned u; } z;
  z.h[0] = (__bf16)a; z.h[1] = (__bf16)b;
  return z.u;
}

// Fragment-order swizzle: chunk (kp-panel, tile t, lane) holds 16 contiguous bf16.
// lane = (row|col) + ((klocal&8)?16:0) ; elem = (klocal&7) + ((klocal&16)?8:0)
__device__ __forceinline__ int sw_dword(int kp, int ntiles, int t, int kl, int rc) {
  int lane = rc + ((kl & 8) ? 16 : 0);
  int e    = (kl & 7) + ((kl & 16) ? 8 : 0);
  return ((kp * ntiles + t) * 32 + lane) * 8 + (e >> 1);   // dword index (pairs of k)
}

__device__ __forceinline__ v16bf frag_ld(const __bf16* sw, int chunk) {
  return *(const v16bf*)(sw + (chunk << 4));
}

// C/D fragment: VGPR i holds (M = i + (lane<16 ? 0 : 8), N = lane&15)
__device__ __forceinline__ void tile_store(float* out, int ld, int lane, v8f c) {
  int col = lane & 15;
  int rb  = (lane & 16) ? 8 : 0;
#pragma unroll
  for (int i = 0; i < 8; ++i) out[(i + rb) * ld + col] = c[i];
}

__device__ __forceinline__ float leaky_(float v) { return v >= 0.f ? v : SLOPE_ * v; }
__device__ __forceinline__ float sigm_(float v)  { return 1.f / (1.f + expf(-v)); }

// ---------------------------------------------------------------- weight pre-swizzle (f32 -> fragment-order bf16)
__global__ void k_prep_w(const float* __restrict__ W, __bf16* __restrict__ dst, int K, int H) {
  int i = blockIdx.x * 256 + threadIdx.x;    // over (K/2)*H
  if (i >= (K >> 1) * H) return;
  int khalf = i / H, c = i - khalf * H;
  int k = khalf << 1;
  float v0 = W[k * H + c], v1 = W[(k + 1) * H + c];
  int kp = k >> 5, kl = k & 31;
  ((unsigned*)dst)[sw_dword(kp, H >> 4, c >> 4, kl, c & 15)] = pk_bf16(v0, v1);
}

// ---------------------------------------------------------------- init
__global__ void k_init(float* accf, int* cnt, int* hist, int* curs) {
  int i = blockIdx.x * 256 + threadIdx.x;
  if (i < 2)   accf[i] = 0.f;
  if (i < 8)   cnt[i]  = 0;
  if (i <= NN) hist[i] = 0;
  if (i < NN)  curs[i] = 0;
}

// ---------------------------------------------------------------- r-MLP (512->256->1)
__global__ __launch_bounds__(256) void k_r_mlp(
    const float* __restrict__ x, const __bf16* __restrict__ Wsw,
    const float* __restrict__ b1, const float* __restrict__ g1,
    const float* __restrict__ bt1, const float* __restrict__ W2,
    const float* __restrict__ b2, const float* __restrict__ u_r,
    float* __restrict__ pr, int* __restrict__ choose, float* accf)
{
  __shared__ __align__(32) char smem[53248];
  __bf16* asw = (__bf16*)smem;             // 16x512 swizzled A (16 KB)
  __bf16* wp  = (__bf16*)(smem + 16384);   // 2 x 16 KB weight panels
  float*  hid = (float*)smem;              // alias asw after GEMM (16 KB)
  float*  prm = (float*)(smem + 49152);    // b1,g,bt,w2 (256 each)
  const int tid = threadIdx.x, wave = tid >> 5, lane = tid & 31;
  const int row0 = blockIdx.x * 16;

  for (int i = tid; i < 8 * DX_; i += 256) {
    int row = i & 15, k = (i >> 4) << 1;
    int gr = row0 + row;
    float v0 = 0.f, v1 = 0.f;
    if (gr < NN) { v0 = x[gr * DX_ + k]; v1 = x[gr * DX_ + k + 1]; }
    ((unsigned*)asw)[sw_dword(k >> 5, 1, 0, k & 31, row)] = pk_bf16(v0, v1);
  }
  if (tid < HR_) { prm[tid] = b1[tid]; prm[HR_ + tid] = g1[tid];
                   prm[2 * HR_ + tid] = bt1[tid]; prm[3 * HR_ + tid] = W2[tid]; }

  const char* wg = (const char*)Wsw;
  const unsigned wpo = ldsoff(wp);
  issue_panel<16384>(wpo, wg, tid);
  v8f c0 = {}, c1 = {};
  for (int kp = 0; kp < 16; ++kp) {
    wait_async_0();          // my slices of panel kp arrived (overlapped with kp-1 compute)
    __syncthreads();         // all slices visible; reads of buffer (kp+1)&1 retired
    if (kp + 1 < 16)
      issue_panel<16384>(wpo + ((kp + 1) & 1) * 16384, wg + (kp + 1) * 16384, tid);
    const __bf16* wpc = wp + (kp & 1) * 8192;
    v16bf fa = frag_ld(asw, kp * 32 + lane);
    v16bf f0 = frag_ld(wpc, (2 * wave) * 32 + lane);
    v16bf f1 = frag_ld(wpc, (2 * wave + 1) * 32 + lane);
    c0 = wmma_bf16(fa, f0, c0);
    c1 = wmma_bf16(fa, f1, c1);
  }
  __syncthreads();
  tile_store(hid + wave * 32, HR_, lane, c0);
  tile_store(hid + wave * 32 + 16, HR_, lane, c1);
  __syncthreads();

  if (tid < 16) {
    int gr = row0 + tid;
    if (gr < NN) {
      float* h = &hid[tid * HR_];
      float mu = 0.f;
      for (int j = 0; j < HR_; ++j) { h[j] += prm[j]; mu += h[j]; }
      mu *= (1.f / HR_);
      float var = 0.f;
      for (int j = 0; j < HR_; ++j) { float d = h[j] - mu; var += d * d; }
      var *= (1.f / HR_);
      float rs = rsqrtf(var + LNEPS_);
      float dot = 0.f;
      for (int j = 0; j < HR_; ++j) {
        float v = (h[j] - mu) * rs * prm[HR_ + j] + prm[2 * HR_ + j];
        dot += leaky_(v) * prm[3 * HR_ + j];
      }
      float p = sigm_(dot + b2[0]);
      pr[gr] = p;
      int ch = (u_r[gr] < p) ? 1 : 0;
      choose[gr] = ch;
      float lp = ch ? logf(p + EPS_) : logf(1.f - p + EPS_);
      float ps = fminf(fmaxf(p, EPS_), 1.f - EPS_);
      float en = -(ps * logf(ps) + (1.f - ps) * logf(1.f - ps));
      atomicAdd(&accf[0], lp);
      atomicAdd(&accf[1], en);
    }
  }
}

// ---------------------------------------------------------------- y-MLP (384->128->256) on ps1 & ps2, fused
__global__ __launch_bounds__(256) void k_y_mlp(
    const float* __restrict__ x, const __bf16* __restrict__ W1sw,
    const float* __restrict__ b1, const float* __restrict__ g1,
    const float* __restrict__ bt1, const __bf16* __restrict__ W2sw,
    const float* __restrict__ b2,
    float* __restrict__ y1, float* __restrict__ y2, float* __restrict__ yagg)
{
  __shared__ __align__(32) char smem[58880];
  __bf16* xs1 = (__bf16*)smem;             // 16x384 swizzled (12 KB)
  __bf16* xs2 = (__bf16*)(smem + 12288);   // 16x384 swizzled
  __bf16* wp  = (__bf16*)(smem + 24576);   // 2 panels, up to 16 KB each
  float*  h1  = (float*)smem;              // alias after layer1 (8 KB)
  float*  h2  = (float*)(smem + 8192);
  __bf16* hb1 = (__bf16*)(smem + 16384);   // 16x128 swizzled (4 KB)
  __bf16* hb2 = (__bf16*)(smem + 20480);
  float*  prm = (float*)(smem + 57344);    // b1,g,bt (128 each)

  const int tid = threadIdx.x, wave = tid >> 5, lane = tid & 31;
  const int row0 = blockIdx.x * 16;

  for (int i = tid; i < 8 * DP_; i += 256) {
    int row = i & 15, k = (i >> 4) << 1;
    int gr = row0 + row;
    float a0 = 0.f, a1 = 0.f, b0 = 0.f, b1v = 0.f;
    if (gr < NN) {
      a0 = x[gr * DX_ + k]; a1 = x[gr * DX_ + k + 1];
      int c2 = (k < 256) ? k : (k + 128);       // ps2 = [x[:, :256] | x[:, 384:512]]
      b0 = x[gr * DX_ + c2]; b1v = x[gr * DX_ + c2 + 1];
    }
    int dw = sw_dword(k >> 5, 1, 0, k & 31, row);
    ((unsigned*)xs1)[dw] = pk_bf16(a0, a1);
    ((unsigned*)xs2)[dw] = pk_bf16(b0, b1v);
  }
  if (tid < HY_) { prm[tid] = b1[tid]; prm[HY_ + tid] = g1[tid]; prm[2 * HY_ + tid] = bt1[tid]; }

  // layer 1: K=384 (12 panels), H=128 (panel = 8 KB)
  const char* wg1 = (const char*)W1sw;
  const unsigned wpo = ldsoff(wp);
  issue_panel<8192>(wpo, wg1, tid);
  v8f a1c = {}, a2c = {};
  for (int kp = 0; kp < 12; ++kp) {
    wait_async_0();
    __syncthreads();
    if (kp + 1 < 12)
      issue_panel<8192>(wpo + ((kp + 1) & 1) * 8192, wg1 + (kp + 1) * 8192, tid);
    const __bf16* wpc = wp + (kp & 1) * 4096;
    v16bf fb = frag_ld(wpc, wave * 32 + lane);
    v16bf f1 = frag_ld(xs1, kp * 32 + lane);
    v16bf f2 = frag_ld(xs2, kp * 32 + lane);
    a1c = wmma_bf16(f1, fb, a1c);
    a2c = wmma_bf16(f2, fb, a2c);
  }
  __syncthreads();
  tile_store(h1 + wave * 16, HY_, lane, a1c);
  tile_store(h2 + wave * 16, HY_, lane, a2c);
  __syncthreads();

  if (tid < 32) {
    int r = tid & 15;
    float*  h  = (tid < 16) ? &h1[r * HY_] : &h2[r * HY_];
    __bf16* hb = (tid < 16) ? hb1 : hb2;
    float mu = 0.f;
    for (int j = 0; j < HY_; ++j) { h[j] += prm[j]; mu += h[j]; }
    mu *= (1.f / HY_);
    float var = 0.f;
    for (int j = 0; j < HY_; ++j) { float d = h[j] - mu; var += d * d; }
    var *= (1.f / HY_);
    float rs = rsqrtf(var + LNEPS_);
    for (int j = 0; j < HY_; ++j) {
      float v = (h[j] - mu) * rs * prm[HY_ + j] + prm[2 * HY_ + j];
      int kl = j & 31;
      int lane2 = r + ((kl & 8) ? 16 : 0);
      int e = (kl & 7) + ((kl & 16) ? 8 : 0);
      hb[((j >> 5) * 32 + lane2) * 16 + e] = (__bf16)leaky_(v);
    }
  }

  // layer 2: K=128 (4 panels), H=256 (panel = 16 KB)
  const char* wg2 = (const char*)W2sw;
  issue_panel<16384>(wpo, wg2, tid);
  v8f o1a = {}, o1b = {}, o2a = {}, o2b = {};
  for (int kp = 0; kp < 4; ++kp) {
    wait_async_0();
    __syncthreads();
    if (kp + 1 < 4)
      issue_panel<16384>(wpo + ((kp + 1) & 1) * 16384, wg2 + (kp + 1) * 16384, tid);
    const __bf16* wpc = wp + (kp & 1) * 8192;
    v16bf fa1 = frag_ld(hb1, kp * 32 + lane);
    v16bf fa2 = frag_ld(hb2, kp * 32 + lane);
    v16bf fb0 = frag_ld(wpc, (2 * wave) * 32 + lane);
    v16bf fb1 = frag_ld(wpc, (2 * wave + 1) * 32 + lane);
    o1a = wmma_bf16(fa1, fb0, o1a);
    o1b = wmma_bf16(fa1, fb1, o1b);
    o2a = wmma_bf16(fa2, fb0, o2a);
    o2b = wmma_bf16(fa2, fb1, o2b);
  }
  int col = lane & 15, rb = (lane & 16) ? 8 : 0;
#pragma unroll
  for (int i = 0; i < 8; ++i) {
    int gr = row0 + i + rb;
    if (gr >= NN) continue;
    int cA = wave * 32 + col, cB = cA + 16;
    float va = o1a[i] + b2[cA], wa = o2a[i] + b2[cA];
    float vb = o1b[i] + b2[cB], wb = o2b[i] + b2[cB];
    y1[gr * DY_ + cA] = va; y2[gr * DY_ + cA] = wa; yagg[gr * DY_ + cA] = leaky_(va + wa);
    y1[gr * DY_ + cB] = vb; y2[gr * DY_ + cB] = wb; yagg[gr * DY_ + cB] = leaky_(vb + wb);
  }
}

// ---------------------------------------------------------------- stable compaction (choose -> Iu / Is)
__global__ __launch_bounds__(256) void k_count(const int* __restrict__ flags, int* __restrict__ bsum) {
  __shared__ int red[256];
  int i = blockIdx.x * 256 + threadIdx.x;
  red[threadIdx.x] = (i < NN && flags[i]) ? 1 : 0;
  __syncthreads();
  for (int s = 128; s > 0; s >>= 1) {
    if (threadIdx.x < s) red[threadIdx.x] += red[threadIdx.x + s];
    __syncthreads();
  }
  if (threadIdx.x == 0) bsum[blockIdx.x] = red[0];
}

__global__ void k_scan_blocks(int* bsum, int nblk, int* total) {
  if (threadIdx.x == 0 && blockIdx.x == 0) {
    int s = 0;
    for (int i = 0; i < nblk; ++i) { int v = bsum[i]; bsum[i] = s; s += v; }
    *total = s;
  }
}

__global__ __launch_bounds__(256) void k_compact(const int* __restrict__ flags,
                                                 const int* __restrict__ bsum,
                                                 int* __restrict__ Iu, int* __restrict__ Is)
{
  __shared__ int woff[8];
  const int tid = threadIdx.x, wave = tid >> 5, lane = tid & 31;
  int i = blockIdx.x * 256 + tid;
  int ch = (i < NN && flags[i]) ? 1 : 0;
  unsigned mask = (unsigned)__ballot(ch);
  if (lane == 0) woff[wave] = __popc(mask);
  __syncthreads();
  if (tid == 0) { int s = 0; for (int w2 = 0; w2 < 8; ++w2) { int v = woff[w2]; woff[w2] = s; s += v; } }
  __syncthreads();
  unsigned lt = (lane == 0) ? 0u : (0xffffffffu >> (32 - lane));
  int cBelow = woff[wave] + __popc(mask & lt);
  int baseU = bsum[blockIdx.x];
  int below = blockIdx.x * 256; if (below > NN) below = NN;
  if (i < NN) {
    if (ch) Iu[baseU + cBelow] = i;
    else    Is[(below - baseU) + (tid - cBelow)] = i;
  }
}

// ---------------------------------------------------------------- ia-MLP (256->128->1) on gathered Iu rows
__global__ __launch_bounds__(256) void k_ia_mlp(
    const float* __restrict__ yagg, const int* __restrict__ Iu,
    const int* __restrict__ cnt, const __bf16* __restrict__ W1sw,
    const float* __restrict__ b1, const float* __restrict__ g1,
    const float* __restrict__ bt1, const float* __restrict__ W2,
    const float* __restrict__ b2, const float* __restrict__ u_ia,
    int* __restrict__ vcN, float* accf)
{
  const int nIu = cnt[0];
  const int row0 = blockIdx.x * 16;
  if (row0 >= nIu) return;
  __shared__ __align__(32) char smem[26624];
  __bf16* asw = (__bf16*)smem;             // 16x256 swizzled (8 KB)
  __bf16* wp  = (__bf16*)(smem + 8192);    // 2 x 8 KB panels
  float*  hid = (float*)smem;              // alias after GEMM (8 KB)
  float*  prm = (float*)(smem + 24576);
  __shared__ int sIu[16];
  const int tid = threadIdx.x, wave = tid >> 5, lane = tid & 31;
  if (tid < 16) sIu[tid] = (row0 + tid < nIu) ? Iu[row0 + tid] : -1;
  if (tid < HY_) { prm[tid] = b1[tid]; prm[HY_ + tid] = g1[tid];
                   prm[2 * HY_ + tid] = bt1[tid]; prm[3 * HY_ + tid] = W2[tid]; }
  __syncthreads();
  for (int i = tid; i < 8 * DY_; i += 256) {
    int row = i & 15, k = (i >> 4) << 1;
    int node = sIu[row];
    float v0 = 0.f, v1 = 0.f;
    if (node >= 0) { v0 = yagg[node * DY_ + k]; v1 = yagg[node * DY_ + k + 1]; }
    ((unsigned*)asw)[sw_dword(k >> 5, 1, 0, k & 31, row)] = pk_bf16(v0, v1);
  }
  const char* wg = (const char*)W1sw;
  const unsigned wpo = ldsoff(wp);
  issue_panel<8192>(wpo, wg, tid);
  v8f acc = {};
  for (int kp = 0; kp < 8; ++kp) {
    wait_async_0();
    __syncthreads();
    if (kp + 1 < 8)
      issue_panel<8192>(wpo + ((kp + 1) & 1) * 8192, wg + (kp + 1) * 8192, tid);
    v16bf fa = frag_ld(asw, kp * 32 + lane);
    v16bf fb = frag_ld(wp + (kp & 1) * 4096, wave * 32 + lane);
    acc = wmma_bf16(fa, fb, acc);
  }
  __syncthreads();
  tile_store(hid + wave * 16, HY_, lane, acc);
  __syncthreads();
  if (tid < 16) {
    int rr = row0 + tid;
    if (rr < nIu) {
      float* h = &hid[tid * HY_];
      float mu = 0.f;
      for (int j = 0; j < HY_; ++j) { h[j] += prm[j]; mu += h[j]; }
      mu *= (1.f / HY_);
      float var = 0.f;
      for (int j = 0; j < HY_; ++j) { float d = h[j] - mu; var += d * d; }
      var *= (1.f / HY_);
      float rs = rsqrtf(var + LNEPS_);
      float dot = 0.f;
      for (int j = 0; j < HY_; ++j) {
        float v = (h[j] - mu) * rs * prm[HY_ + j] + prm[2 * HY_ + j];
        dot += leaky_(v) * prm[3 * HY_ + j];
      }
      float p = sigm_(dot + b2[0]);
      int node = sIu[tid];
      int v = (u_ia[node] < p) ? 1 : 0;
      vcN[node] = v;
      float lp = v ? logf(p + EPS_) : logf(1.f - p + EPS_);
      float ps = fminf(fmaxf(p, EPS_), 1.f - EPS_);
      float en = -(ps * logf(ps) + (1.f - ps) * logf(1.f - ps));
      atomicAdd(&accf[0], lp);
      atomicAdd(&accf[1], en);
    }
  }
}

// ---------------------------------------------------------------- edge pipeline
__global__ void k_keep(const int* __restrict__ choose, const int* __restrict__ vcN,
                       int* __restrict__ keep) {
  int i = blockIdx.x * 256 + threadIdx.x;
  if (i < NN) keep[i] = (choose[i] && !vcN[i]) ? 1 : 0;
}

__global__ void k_edge_hist(const long long* __restrict__ ei, const int* __restrict__ keep,
                            int* __restrict__ hist) {
  int idx = blockIdx.x * 256 + threadIdx.x;
  if (idx >= E2_) return;
  int e = (idx < M_) ? idx : idx - M_;
  int par = (idx < M_) ? (int)ei[e] : (int)ei[M_ + e];
  if (keep[par]) atomicAdd(&hist[par], 1);
}

__global__ void k_scan_hist(const int* __restrict__ hist, int* __restrict__ segb, int* cnt) {
  if (threadIdx.x == 0 && blockIdx.x == 0) {
    int s = 0;
    for (int i = 0; i < NN; ++i) { segb[i] = s; s += hist[i]; }
    segb[NN] = s;
    cnt[1] = s;
  }
}

__global__ void k_edge_scatter(const long long* __restrict__ ei, const int* __restrict__ keep,
                               const int* __restrict__ segb, int* __restrict__ curs,
                               int* __restrict__ PAR, int* __restrict__ NBE, int* __restrict__ EKE)
{
  int idx = blockIdx.x * 256 + threadIdx.x;
  if (idx >= E2_) return;
  int e = (idx < M_) ? idx : idx - M_;
  int par, nb;
  if (idx < M_) { par = (int)ei[e];      nb = (int)ei[M_ + e]; }
  else          { par = (int)ei[M_ + e]; nb = (int)ei[e]; }
  if (!keep[par]) return;
  int pos = segb[par] + atomicAdd(&curs[par], 1);
  PAR[pos] = par; NBE[pos] = nb; EKE[pos] = e;
}

__global__ void k_seg_sort(const int* __restrict__ hist, const int* __restrict__ segb,
                           int* __restrict__ NBE, int* __restrict__ EKE)
{
  int s = blockIdx.x * 256 + threadIdx.x;
  if (s >= NN) return;
  int c = hist[s];
  if (c < 2) return;
  int base = segb[s];
  for (int a = base + 1; a < base + c; ++a) {
    int kn = NBE[a], ke = EKE[a];
    int b = a - 1;
    while (b >= base && (NBE[b] > kn || (NBE[b] == kn && EKE[b] > ke))) {
      NBE[b + 1] = NBE[b]; EKE[b + 1] = EKE[b]; --b;
    }
    NBE[b + 1] = kn; EKE[b + 1] = ke;
  }
}

// ---------------------------------------------------------------- c-MLP (832->128->1)
__global__ __launch_bounds__(256) void k_c_mlp(
    const float* __restrict__ x, const float* __restrict__ ea,
    const float* __restrict__ yagg,
    const int* __restrict__ PAR, const int* __restrict__ NBE, const int* __restrict__ EKE,
    const int* __restrict__ cnt, const __bf16* __restrict__ W1sw,
    const float* __restrict__ b1, const float* __restrict__ g1,
    const float* __restrict__ bt1, const float* __restrict__ W2,
    const float* __restrict__ b2, float* __restrict__ score)
{
  const int Pn = cnt[1];
  const int p0 = blockIdx.x * 16;
  if (p0 >= Pn) return;
  __shared__ __align__(32) char smem[53440];
  __bf16* fsw = (__bf16*)smem;             // 16x832 swizzled (26624 B)
  __bf16* wp  = (__bf16*)(smem + 26624);   // 2 x 8 KB
  float*  hid = (float*)(smem + 43008);    // 8 KB
  float*  prm = (float*)(smem + 51200);
  int* sp = (int*)(smem + 53248);
  int* sn = sp + 16; int* se = sp + 32;
  const int tid = threadIdx.x, wave = tid >> 5, lane = tid & 31;
  if (tid < 16) {
    int p = p0 + tid;
    sp[tid] = (p < Pn) ? PAR[p] : -1;
    sn[tid] = (p < Pn) ? NBE[p] : 0;
    se[tid] = (p < Pn) ? EKE[p] : 0;
  }
  if (tid < HY_) { prm[tid] = b1[tid]; prm[HY_ + tid] = g1[tid];
                   prm[2 * HY_ + tid] = bt1[tid]; prm[3 * HY_ + tid] = W2[tid]; }
  __syncthreads();
  for (int i = tid; i < 8 * DF_; i += 256) {
    int row = i & 15, k = (i >> 4) << 1;
    float v0 = 0.f, v1 = 0.f;
    if (sp[row] >= 0) {
      if (k < DY_)            { v0 = yagg[sp[row] * DY_ + k];        v1 = yagg[sp[row] * DY_ + k + 1]; }
      else if (k < DY_ + DW_) { v0 = ea[se[row] * DW_ + (k - DY_)];  v1 = ea[se[row] * DW_ + (k - DY_) + 1]; }
      else                    { int cc = k - DY_ - DW_;
                                v0 = x[sn[row] * DX_ + cc];          v1 = x[sn[row] * DX_ + cc + 1]; }
    }
    ((unsigned*)fsw)[sw_dword(k >> 5, 1, 0, k & 31, row)] = pk_bf16(v0, v1);
  }
  const char* wg = (const char*)W1sw;
  const unsigned wpo = ldsoff(wp);
  issue_panel<8192>(wpo, wg, tid);
  v8f acc = {};
  for (int kp = 0; kp < 26; ++kp) {
    wait_async_0();
    __syncthreads();
    if (kp + 1 < 26)
      issue_panel<8192>(wpo + ((kp + 1) & 1) * 8192, wg + (kp + 1) * 8192, tid);
    v16bf fa = frag_ld(fsw, kp * 32 + lane);
    v16bf fb = frag_ld(wp + (kp & 1) * 4096, wave * 32 + lane);
    acc = wmma_bf16(fa, fb, acc);
  }
  __syncthreads();
  tile_store(hid + wave * 16, HY_, lane, acc);
  __syncthreads();
  if (tid < 16) {
    int p = p0 + tid;
    if (p < Pn) {
      float* h = &hid[tid * HY_];
      float mu = 0.f;
      for (int j = 0; j < HY_; ++j) { h[j] += prm[j]; mu += h[j]; }
      mu *= (1.f / HY_);
      float var = 0.f;
      for (int j = 0; j < HY_; ++j) { float d = h[j] - mu; var += d * d; }
      var *= (1.f / HY_);
      float rs = rsqrtf(var + LNEPS_);
      float dot = 0.f;
      for (int j = 0; j < HY_; ++j) {
        float v = (h[j] - mu) * rs * prm[HY_ + j] + prm[2 * HY_ + j];
        dot += leaky_(v) * prm[3 * HY_ + j];
      }
      score[p] = dot + b2[0];
    }
  }
}

// ---------------------------------------------------------------- segmented softmax + pick + logP/ent
__global__ void k_softmax_pick(const int* __restrict__ hist, const int* __restrict__ segb,
                               const float* __restrict__ score, const float* __restrict__ u_pick,
                               float* accf)
{
  int s = blockIdx.x * 256 + threadIdx.x;
  if (s >= NN) return;
  int c = hist[s];
  if (c == 0) return;
  int base = segb[s];
  float mx = -3.4e38f;
  for (int j = 0; j < c; ++j) mx = fmaxf(mx, score[base + j]);
  float den = 0.f;
  for (int j = 0; j < c; ++j) den += expf(score[base + j] - mx);
  float u = u_pick[s];
  float cum = 0.f, ent = 0.f;
  int pick = c - 1, found = 0;
  for (int j = 0; j < c; ++j) {
    float prb = expf(score[base + j] - mx) / den;
    float pc = fmaxf(prb, EPS_);
    ent -= pc * logf(pc);
    cum += prb;
    if (!found && cum > u) { pick = j; found = 1; }
  }
  float pp = expf(score[base + pick] - mx) / den;
  atomicAdd(&accf[0], logf(fmaxf(pp, EPS_)));
  atomicAdd(&accf[1], ent);
}

// ---------------------------------------------------------------- output assembly
__global__ __launch_bounds__(256) void k_xout(
    const float* __restrict__ y1, const float* __restrict__ y2,
    const int* __restrict__ Iu, const int* __restrict__ Is,
    const int* __restrict__ cnt, float* __restrict__ out)
{
  int nIu = cnt[0];
  int nIs = NN - nIu;
  int row = blockIdx.x;
  if (row >= nIs + 2 * nIu) return;
  const float* src;
  if (row < nIs)            src = &y1[Is[row] * DY_];
  else if (row < nIs + nIu) src = &y1[Iu[row - nIs] * DY_];
  else                      src = &y2[Iu[row - nIs - nIu] * DY_];
  out[(size_t)row * DY_ + threadIdx.x] = src[threadIdx.x];
}

__global__ void k_final(const float* accf, const int* cnt, float* out) {
  if (threadIdx.x == 0 && blockIdx.x == 0) {
    size_t off = (size_t)(NN + cnt[0]) * DY_;
    out[off]     = accf[0];
    out[off + 1] = accf[1];
  }
}

// ---------------------------------------------------------------- launch
extern "C" void kernel_launch(void* const* d_in, const int* in_sizes, int n_in,
                              void* d_out, int out_size, void* d_ws, size_t ws_size,
                              hipStream_t stream)
{
  (void)in_sizes; (void)n_in; (void)out_size; (void)ws_size;
  const float*     x      = (const float*)d_in[0];
  const long long* ei     = (const long long*)d_in[1];
  const float*     ea     = (const float*)d_in[2];
  const float*     u_r    = (const float*)d_in[3];
  const float*     u_ia   = (const float*)d_in[4];
  const float*     u_pick = (const float*)d_in[5];
  const float *rW1 = (const float*)d_in[6],  *rb1  = (const float*)d_in[7],
              *rg1 = (const float*)d_in[8],  *rbt1 = (const float*)d_in[9],
              *rW2 = (const float*)d_in[10], *rb2  = (const float*)d_in[11];
  const float *yW1 = (const float*)d_in[12], *yb1  = (const float*)d_in[13],
              *yg1 = (const float*)d_in[14], *ybt1 = (const float*)d_in[15],
              *yW2 = (const float*)d_in[16], *yb2  = (const float*)d_in[17];
  const float *iaW1 = (const float*)d_in[18], *iab1  = (const float*)d_in[19],
              *iag1 = (const float*)d_in[20], *iabt1 = (const float*)d_in[21],
              *iaW2 = (const float*)d_in[22], *iab2  = (const float*)d_in[23];
  const float *cW1 = (const float*)d_in[24], *cb1  = (const float*)d_in[25],
              *cg1 = (const float*)d_in[26], *cbt1 = (const float*)d_in[27],
              *cW2 = (const float*)d_in[28], *cb2  = (const float*)d_in[29];

  char* w = (char*)d_ws;
  float* accf   = (float*)(w + 0);
  int*   cnt    = (int*)(w + 256);
  int*   bsum   = (int*)(w + 512);
  float* pr     = (float*)(w + 4608);
  int*   choose = (int*)(w + 84608);
  int*   Iu     = (int*)(w + 164608);
  int*   Is     = (int*)(w + 244608);
  int*   vcN    = (int*)(w + 404608);
  int*   keep   = (int*)(w + 484608);
  int*   hist   = (int*)(w + 564608);
  int*   curs   = (int*)(w + 644864);
  int*   segb   = (int*)(w + 724864);
  float* y1     = (float*)(w + 805120);
  float* y2     = (float*)(w + 805120 + 20480000);
  float* yagg   = (float*)(w + 805120 + 2 * 20480000);
  int*   PARa   = (int*)(w + 62245120);
  int*   NBEa   = (int*)(w + 64805120);
  int*   EKEa   = (int*)(w + 67365120);
  float* score  = (float*)(w + 69925120);
  __bf16* rW1sw  = (__bf16*)(w + 72485120);   // 512x256 -> 262144 B
  __bf16* yW1sw  = (__bf16*)(w + 72747264);   // 384x128 ->  98304 B
  __bf16* yW2sw  = (__bf16*)(w + 72845568);   // 128x256 ->  65536 B
  __bf16* iaW1sw = (__bf16*)(w + 72911104);   // 256x128 ->  65536 B
  __bf16* cW1sw  = (__bf16*)(w + 72976640);   // 832x128 -> 212992 B

  float* out = (float*)d_out;
  dim3 b256(256);

  k_init<<<NB256, b256, 0, stream>>>(accf, cnt, hist, curs);
  k_prep_w<<<(256 * 256 + 255) / 256, b256, 0, stream>>>(rW1, rW1sw, DX_, HR_);
  k_prep_w<<<(192 * 128 + 255) / 256, b256, 0, stream>>>(yW1, yW1sw, DP_, HY_);
  k_prep_w<<<(64 * 256 + 255) / 256, b256, 0, stream>>>(yW2, yW2sw, HY_, DY_);
  k_prep_w<<<(128 * 128 + 255) / 256, b256, 0, stream>>>(iaW1, iaW1sw, DY_, HY_);
  k_prep_w<<<(416 * 128 + 255) / 256, b256, 0, stream>>>(cW1, cW1sw, DF_, HY_);

  k_r_mlp<<<(NN + 15) / 16, b256, 0, stream>>>(x, rW1sw, rb1, rg1, rbt1, rW2, rb2,
                                               u_r, pr, choose, accf);
  k_y_mlp<<<(NN + 15) / 16, b256, 0, stream>>>(x, yW1sw, yb1, yg1, ybt1, yW2sw, yb2,
                                               y1, y2, yagg);
  k_count<<<NB256, b256, 0, stream>>>(choose, bsum);
  k_scan_blocks<<<1, 1, 0, stream>>>(bsum, NB256, &cnt[0]);
  k_compact<<<NB256, b256, 0, stream>>>(choose, bsum, Iu, Is);
  k_ia_mlp<<<(NN + 15) / 16, b256, 0, stream>>>(yagg, Iu, cnt, iaW1sw, iab1, iag1,
                                                iabt1, iaW2, iab2, u_ia, vcN, accf);
  k_keep<<<NB256, b256, 0, stream>>>(choose, vcN, keep);
  k_edge_hist<<<(E2_ + 255) / 256, b256, 0, stream>>>(ei, keep, hist);
  k_scan_hist<<<1, 1, 0, stream>>>(hist, segb, cnt);
  k_edge_scatter<<<(E2_ + 255) / 256, b256, 0, stream>>>(ei, keep, segb, curs,
                                                         PARa, NBEa, EKEa);
  k_seg_sort<<<NB256, b256, 0, stream>>>(hist, segb, NBEa, EKEa);
  k_c_mlp<<<(E2_ + 15) / 16, b256, 0, stream>>>(x, ea, yagg, PARa, NBEa, EKEa, cnt,
                                                cW1sw, cb1, cg1, cbt1, cW2, cb2, score);
  k_softmax_pick<<<NB256, b256, 0, stream>>>(hist, segb, score, u_pick, accf);
  k_xout<<<2 * NN, b256, 0, stream>>>(y1, y2, Iu, Is, cnt, out);
  k_final<<<1, 1, 0, stream>>>(accf, cnt, out);
}
